// GNNEncoder_45938970198548
// MI455X (gfx1250) — matrix operand order repaired and checked by reading, not compiled, not run on hardware
//
#include <hip/hip_runtime.h>
#include <hip/hip_bf16.h>

typedef __attribute__((ext_vector_type(16))) __bf16 v16bf;
typedef __attribute__((ext_vector_type(8)))  float  v8f;

typedef __attribute__((address_space(1))) int g1_int;   // global (AS1) int
typedef __attribute__((address_space(3))) int l3_int;   // LDS (AS3) int

union ABFrag { unsigned int u[8]; uint4 q[2]; v16bf v; };

__device__ __forceinline__ unsigned short f2bf(float x) {
  unsigned int u = __float_as_uint(x);
  u += 0x7FFFu + ((u >> 16) & 1u);          // round-to-nearest-even
  return (unsigned short)(u >> 16);
}

// ---------------- degree / norm ----------------
__global__ void k_deg_init(float* __restrict__ deg, int n) {
  int i = blockIdx.x * 256 + threadIdx.x;
  if (i < n) deg[i] = 1.0f;                 // self-loop
}
__global__ void k_deg_count(const long long* __restrict__ ei, float* __restrict__ deg, int e) {
  int i = blockIdx.x * 256 + threadIdx.x;
  if (i < e) atomicAdd(&deg[(int)ei[(long long)e + i]], 1.0f);
}
__global__ void k_deg_rsqrt(float* __restrict__ deg, int n) {
  int i = blockIdx.x * 256 + threadIdx.x;
  if (i < n) deg[i] = rsqrtf(deg[i]);
}

// ---------------- fp32 -> bf16 conversion ----------------
__global__ void k_cvt_bf16(const float* __restrict__ x, unsigned short* __restrict__ y, long long n) {
  long long i = (long long)blockIdx.x * 256 + threadIdx.x;
  long long stride = (long long)gridDim.x * 256;
  for (; i < n; i += stride) y[i] = f2bf(x[i]);
}

// ---- pack W (K x Nc, row-major fp32) into WMMA B-operand lane layout ----
// dword idx = ((tn*(K/32)+kc)*32 + lane)*8 + d ; lane covers col=tn*16+(l&15),
// K pair = kc*32 + (l>>4)*16 + 2d  (B 32x16 bf16 as packed here: lanes0-15 K=0..15, lanes16-31 K=16..31)
__global__ void k_pack_w(const float* __restrict__ W, unsigned int* __restrict__ Wp, int K, int Nc) {
  long long total = (long long)K * Nc / 2;
  long long idx = (long long)blockIdx.x * 256 + threadIdx.x;
  if (idx >= total) return;
  int d  = (int)(idx & 7);
  int l  = (int)((idx >> 3) & 31);
  int KC = K >> 5;
  int kc = (int)((idx >> 8) % KC);
  int tn = (int)((idx >> 8) / KC);
  int col = tn * 16 + (l & 15);
  int k0  = kc * 32 + ((l >> 4) * 16) + 2 * d;
  unsigned int lo = f2bf(W[(long long)k0 * Nc + col]);
  unsigned int hi = f2bf(W[(long long)(k0 + 1) * Nc + col]);
  Wp[idx] = lo | (hi << 16);
}

// ---------------- WMMA GEMM: H(MxNc) = Abf(MxK) * Wpack ----------------
// one block = 16 rows x 256 cols; 8 waves, wave w owns n-tiles w and w+8.
// A tile (16x32 bf16 = 1KB) is double-buffered in LDS and prefetched with
// GLOBAL_LOAD_ASYNC_TO_LDS_B32 (ASYNCcnt) one k-step ahead: one barrier/iter.
__global__ void __launch_bounds__(256) k_gemm_bf16(const unsigned short* __restrict__ Abf,
                                                   const unsigned int*  __restrict__ Wpack,
                                                   float* __restrict__ H, int M, int K, int Nc) {
  __shared__ unsigned int lds_a[512];       // 2 x (16 rows x 16 dwords)
  const int tid  = threadIdx.x;
  const int lane = tid & 31;
  const int wv   = tid >> 5;
  const int mrow0 = blockIdx.x * 16;
  const int KC = K >> 5;
  const int Kd = K >> 1;                    // dwords per A row
  const unsigned int* A32 = (const unsigned int*)Abf;

  const int row = lane & 15;
  const int hh  = lane >> 4;
  const int abase = row * 16 + hh * 8;
  const int tn0 = wv, tn1 = wv + 8;
  const int ldr = tid >> 4, ldc = tid & 15; // cooperative A-tile load coords
  const long long arowoff = (long long)(mrow0 + ldr) * Kd + ldc;

  // prologue: async-prefetch tile kc=0 into buffer 0 (one b32 per thread)
  __builtin_amdgcn_global_load_async_to_lds_b32(
      (g1_int*)(A32 + arowoff), (l3_int*)&lds_a[tid], 0, 0);

  v8f acc0 = {}; v8f acc1 = {};
  for (int kc = 0; kc < KC; ++kc) {
    // B fragments: independent of LDS, issue before the sync to hide latency
    ABFrag bf0, bf1;
    const uint4* bp0 = (const uint4*)(Wpack + ((long long)(tn0 * KC + kc) * 32 + lane) * 8);
    const uint4* bp1 = (const uint4*)(Wpack + ((long long)(tn1 * KC + kc) * 32 + lane) * 8);
    bf0.q[0] = bp0[0]; bf0.q[1] = bp0[1];
    bf1.q[0] = bp1[0]; bf1.q[1] = bp1[1];

    // make tile kc visible: own async chunk done, then workgroup-wide barrier
    asm volatile("s_wait_asynccnt 0x0" ::: "memory");
    __syncthreads();

    // prefetch tile kc+1 into the other buffer (readers of it finished before
    // the barrier above: their ds_loads complete before their WMMA issues)
    if (kc + 1 < KC) {
      __builtin_amdgcn_global_load_async_to_lds_b32(
          (g1_int*)(A32 + arowoff + (long long)(kc + 1) * 16),
          (l3_int*)&lds_a[((kc + 1) & 1) * 256 + tid], 0, 0);
    }

    ABFrag af;
    const unsigned int* lp = &lds_a[(kc & 1) * 256 + abase];
    af.q[0] = *(const uint4*)(lp);
    af.q[1] = *(const uint4*)(lp + 4);
    acc0 = __builtin_amdgcn_wmma_f32_16x16x32_bf16(false, af.v, false, bf0.v, (short)0, acc0, false, false);
    acc1 = __builtin_amdgcn_wmma_f32_16x16x32_bf16(false, af.v, false, bf1.v, (short)0, acc1, false, false);
  }
  const int col0 = tn0 * 16 + row;
  const int col1 = tn1 * 16 + row;
#pragma unroll
  for (int d = 0; d < 8; ++d) {
    int m = mrow0 + hh * 8 + d;             // C/D layout: VGPR d -> rows d (lo half) / d+8 (hi half)
    H[(long long)m * Nc + col0] = acc0[d];
    H[(long long)m * Nc + col1] = acc1[d];
  }
}

// ---------------- aggregation ----------------
__global__ void k_selfloop_init(const float* __restrict__ H, const float* __restrict__ dinv,
                                float* __restrict__ agg, int n) {
  int r = blockIdx.x;
  if (r >= n) return;
  float w = dinv[r]; w *= w;
  long long o = (long long)r * 256 + threadIdx.x;
  agg[o] = H[o] * w;
}

__global__ void k_scatter(const long long* __restrict__ ei, const float* __restrict__ H,
                          const float* __restrict__ dinv, float* __restrict__ agg, int e) {
  int t = threadIdx.x;
  long long edge = (long long)blockIdx.x * 4 + (t >> 6);
  if (edge >= e) return;
  int s  = (int)ei[edge];
  int dt = (int)ei[(long long)e + edge];
  float w = dinv[s] * dinv[dt];
  int f = (t & 63) * 4;
  float4 h4 = *(const float4*)&H[(long long)s * 256 + f];
  float* o = &agg[(long long)dt * 256 + f];
  atomicAdd(o + 0, h4.x * w);
  atomicAdd(o + 1, h4.y * w);
  atomicAdd(o + 2, h4.z * w);
  atomicAdd(o + 3, h4.w * w);
}

// ---------------- BatchNorm (+ReLU) ----------------
__global__ void k_zero(float* __restrict__ p, int n) {
  int i = blockIdx.x * 256 + threadIdx.x;
  if (i < n) p[i] = 0.0f;
}
__global__ void k_bn_stats(const float* __restrict__ agg, float* __restrict__ gsum,
                           float* __restrict__ gssq, int n) {
  int f = threadIdx.x;                      // 256 features
  float s = 0.f, ss = 0.f;
  for (long long r = blockIdx.x; r < n; r += gridDim.x) {
    float v = agg[r * 256 + f];
    s += v; ss += v * v;
  }
  atomicAdd(&gsum[f], s);
  atomicAdd(&gssq[f], ss);
}
__global__ void k_bn_finalize(const float* __restrict__ gsum, const float* __restrict__ gssq,
                              const float* __restrict__ gamma, const float* __restrict__ beta,
                              float* __restrict__ scale, float* __restrict__ shift, int n) {
  int f = threadIdx.x;
  float inv_n = 1.0f / (float)n;
  float mean = gsum[f] * inv_n;
  float var  = gssq[f] * inv_n - mean * mean;
  var = var > 0.f ? var : 0.f;
  float sc = gamma[f] * rsqrtf(var + 1e-5f);
  scale[f] = sc;
  shift[f] = beta[f] - mean * sc;           // GCN bias cancels through batch-mean subtraction
}
__global__ void k_bn_relu_apply(const float* __restrict__ agg, const float* __restrict__ scale,
                                const float* __restrict__ shift, float* __restrict__ yf,
                                unsigned short* __restrict__ ybf, int n) {
  int r = blockIdx.x;
  if (r >= n) return;
  int f = threadIdx.x;
  long long o = (long long)r * 256 + f;
  float v = fmaf(agg[o], scale[f], shift[f]);
  v = v > 0.f ? v : 0.f;
  if (yf)  yf[o]  = v;
  if (ybf) ybf[o] = f2bf(v);
}

extern "C" void kernel_launch(void* const* d_in, const int* in_sizes, int n_in,
                              void* d_out, int out_size, void* d_ws, size_t ws_size,
                              hipStream_t stream) {
  const float*     X      = (const float*)d_in[0];
  const long long* ei     = (const long long*)d_in[1];
  const float*     W1     = (const float*)d_in[2];
  const float*     gamma1 = (const float*)d_in[4];
  const float*     beta1  = (const float*)d_in[5];
  const float*     W2     = (const float*)d_in[6];
  const float*     gamma2 = (const float*)d_in[8];
  const float*     beta2  = (const float*)d_in[9];
  float* out = (float*)d_out;

  const int HID = in_sizes[3];              // 256
  const int K1  = in_sizes[2] / HID;        // 1024
  const int N   = in_sizes[0] / K1;         // 50000
  const int E   = in_sizes[1] / 2;          // 800000

  // workspace layout
  char* w = (char*)d_ws;
  size_t off = 0;
  auto alloc = [&](size_t bytes) -> void* {
    void* p = w + off;
    off = (off + bytes + 255) & ~(size_t)255;
    return p;
  };
  unsigned short* Xbf   = (unsigned short*)alloc((size_t)N * K1 * 2);
  unsigned int*   W1p   = (unsigned int*)  alloc((size_t)K1 * HID * 2);
  unsigned int*   W2p   = (unsigned int*)  alloc((size_t)HID * HID * 2);
  float*          Hbuf  = (float*)         alloc((size_t)N * HID * 4);
  float*          agg   = (float*)         alloc((size_t)N * HID * 4);
  unsigned short* h1bf  = (unsigned short*)alloc((size_t)N * HID * 2);
  float*          dinv  = (float*)         alloc((size_t)N * 4);
  float*          gsum  = (float*)         alloc(256 * 4);
  float*          gssq  = (float*)         alloc(256 * 4);
  float*          scale = (float*)         alloc(256 * 4);
  float*          shift = (float*)         alloc(256 * 4);
  (void)ws_size; (void)n_in; (void)out_size;

  const int nb_n = (N + 255) / 256;
  const int nb_e = (E + 255) / 256;

  // normalization coefficients
  k_deg_init <<<nb_n, 256, 0, stream>>>(dinv, N);
  k_deg_count<<<nb_e, 256, 0, stream>>>(ei, dinv, E);
  k_deg_rsqrt<<<nb_n, 256, 0, stream>>>(dinv, N);

  // bf16 inputs + packed weights
  k_cvt_bf16<<<8192, 256, 0, stream>>>(X, Xbf, (long long)N * K1);
  {
    long long t1 = (long long)K1 * HID / 2;
    long long t2 = (long long)HID * HID / 2;
    k_pack_w<<<(int)((t1 + 255) / 256), 256, 0, stream>>>(W1, W1p, K1, HID);
    k_pack_w<<<(int)((t2 + 255) / 256), 256, 0, stream>>>(W2, W2p, HID, HID);
  }

  // ---- layer 1 ----
  k_gemm_bf16<<<N / 16, 256, 0, stream>>>(Xbf, W1p, Hbuf, N, K1, HID);
  k_selfloop_init<<<N, 256, 0, stream>>>(Hbuf, dinv, agg, N);
  k_scatter<<<(E + 3) / 4, 256, 0, stream>>>(ei, Hbuf, dinv, agg, E);
  k_zero<<<2, 256, 0, stream>>>(gsum, 256);
  k_zero<<<2, 256, 0, stream>>>(gssq, 256);
  k_bn_stats<<<1024, 256, 0, stream>>>(agg, gsum, gssq, N);
  k_bn_finalize<<<1, 256, 0, stream>>>(gsum, gssq, gamma1, beta1, scale, shift, N);
  k_bn_relu_apply<<<N, 256, 0, stream>>>(agg, scale, shift, nullptr, h1bf, N);

  // ---- layer 2 ----
  k_gemm_bf16<<<N / 16, 256, 0, stream>>>(h1bf, W2p, Hbuf, N, HID, HID);
  k_selfloop_init<<<N, 256, 0, stream>>>(Hbuf, dinv, agg, N);
  k_scatter<<<(E + 3) / 4, 256, 0, stream>>>(ei, Hbuf, dinv, agg, E);
  k_zero<<<2, 256, 0, stream>>>(gsum, 256);
  k_zero<<<2, 256, 0, stream>>>(gssq, 256);
  k_bn_stats<<<1024, 256, 0, stream>>>(agg, gsum, gssq, N);
  k_bn_finalize<<<1, 256, 0, stream>>>(gsum, gssq, gamma2, beta2, scale, shift, N);
  k_bn_relu_apply<<<N, 256, 0, stream>>>(agg, scale, shift, out, nullptr, N);
}
